// MultiHeadAttention_27315992003308
// MI455X (gfx1250) — matrix operand order repaired and checked by reading, not compiled, not run on hardware
//
#include <hip/hip_runtime.h>
#include <hip/hip_bf16.h>

// MHA forward for MI455X (gfx1250, wave32, WMMA f16 16x16x32, f32 accumulate).
// Workspace usage (f16 elements):
//   Qh [32][2048][64] row-major      : 4M halves (8 MB)
//   Kh [32][2048][64] row-major      : 4M halves (8 MB)
//   Vt [32][64][2048] transposed     : 4M halves (8 MB)
//   Xq (queries f16)                 : 4M halves (8 MB)
//   Xk (keys f16)                    : 4M halves (8 MB)
//   Wh (Wq,Wk,Wv f16)                : 3M halves (6 MB)
// Total: 46 MB.

#define B_   2
#define T_   2048
#define D_   1024
#define H_   16
#define DK_  64
#define BH_  (B_*H_)

typedef _Float16 v4h  __attribute__((ext_vector_type(4)));
typedef _Float16 v8h  __attribute__((ext_vector_type(8)));
typedef _Float16 v16h __attribute__((ext_vector_type(16)));
typedef float    v8f  __attribute__((ext_vector_type(8)));

#define WMMA_F16(a, b, c) \
  __builtin_amdgcn_wmma_f32_16x16x32_f16(false, (a), false, (b), (short)0, (c), false, false)

static __device__ __forceinline__ v16h cat8(v8h lo, v8h hi) {
  return __builtin_shufflevector(lo, hi, 0,1,2,3,4,5,6,7,8,9,10,11,12,13,14,15);
}

// A-fragment: 16(M) x 32(K) f16, source row-major [M][K] with row stride ld.
// Contiguous k order (same as B): lanes 0-15 hold M=lane, K=0..15; lanes 16-31 hold
// M=lane-16, K=16..31. Two 16B loads per lane.
static __device__ __forceinline__ v16h load_a_frag(const _Float16* base, int ld, int lane) {
  const _Float16* p = base + (size_t)(lane & 15) * ld + ((lane < 16) ? 0 : 16);
  v8h lo = *(const v8h*)p;
  v8h hi = *(const v8h*)(p + 8);
  return cat8(lo, hi);
}

// B-fragment: 32(K) x 16(N) f16, source stored [N][K] row-major, row stride ld.
// Contiguous k order (same as A): lanes 0-15 hold N=lane, K=0..15; lanes 16-31 hold N=lane-16,
// K=16..31. Two 16B loads per lane.
static __device__ __forceinline__ v16h load_b_frag(const _Float16* base, int ld, int lane) {
  const _Float16* p = base + (size_t)(lane & 15) * ld + ((lane < 16) ? 0 : 16);
  v8h lo = *(const v8h*)p;
  v8h hi = *(const v8h*)(p + 8);
  return cat8(lo, hi);
}

// ---------------------------------------------------------------------------
// Kernel 0: f32 -> f16 conversion (vectorized x4).
// ---------------------------------------------------------------------------
__global__ void mha_cvt_f32f16(const float* __restrict__ src,
                               _Float16* __restrict__ dst, int n4) {
  int i = blockIdx.x * blockDim.x + threadIdx.x;
  if (i >= n4) return;
  float4 v = ((const float4*)src)[i];
  v4h h;
  h[0] = (_Float16)v.x; h[1] = (_Float16)v.y;
  h[2] = (_Float16)v.z; h[3] = (_Float16)v.w;
  ((v4h*)dst)[i] = h;
}

// ---------------------------------------------------------------------------
// Kernel 1: projection GEMM  out[n][c] = sum_k X[n][k] * W[c][k] + bias[c]
// X: [4096][1024] f16 row-major, W: [1024][1024] f16 row-major ([out][in], i.e.
// already [N][K] for the B-fragment). Block = 128 thr (4 waves), tile 64Mx64N,
// each wave owns 16 M-rows and 4 N-subtiles. K-loop: 32 steps of 32, hand
// unrolled x2 with ping-pong fragment buffers: next-step loads stay in flight
// across the current WMMA group (partial s_wait_loadcnt) and no register
// rotation copies are needed.
// is_v == 0: write row-major [bh][t][dk]; is_v == 1: write transposed
// [bh][dk][t] (contiguous per-lane v8h stores thanks to the C-frag layout).
// ---------------------------------------------------------------------------
__global__ __launch_bounds__(128)
void mha_qkv_proj(const _Float16* __restrict__ X, const _Float16* __restrict__ W,
                  const float* __restrict__ bias, _Float16* __restrict__ outRM,
                  _Float16* __restrict__ outTR, int is_v) {
  const int lane  = threadIdx.x & 31;
  const int wave  = threadIdx.x >> 5;
  const int m0    = blockIdx.x * 64 + wave * 16;
  const int ctile = blockIdx.y * 64;

  v8f zero = {0.f, 0.f, 0.f, 0.f, 0.f, 0.f, 0.f, 0.f};
  v8f acc[4] = {zero, zero, zero, zero};

  const _Float16* xrow = X + (size_t)m0 * D_;
  const _Float16* wrow = W + (size_t)ctile * D_;

  v16h a0, a1, b0[4], b1[4];

  // prime buffer 0 with k-step 0
  a0 = load_a_frag(xrow, D_, lane);
#pragma unroll
  for (int nt = 0; nt < 4; ++nt)
    b0[nt] = load_b_frag(wrow + (size_t)nt * 16 * D_, D_, lane);

  // 15 iterations cover k-steps 0..29; each half loads the other buffer.
  for (int i = 0; i < 15; ++i) {
    const int k1 = (2 * i + 1) * 32;
    const int k2 = (2 * i + 2) * 32;

    a1 = load_a_frag(xrow + k1, D_, lane);
#pragma unroll
    for (int nt = 0; nt < 4; ++nt)
      b1[nt] = load_b_frag(wrow + (size_t)nt * 16 * D_ + k1, D_, lane);
#pragma unroll
    for (int nt = 0; nt < 4; ++nt)
      acc[nt] = WMMA_F16(a0, b0[nt], acc[nt]);

    a0 = load_a_frag(xrow + k2, D_, lane);
#pragma unroll
    for (int nt = 0; nt < 4; ++nt)
      b0[nt] = load_b_frag(wrow + (size_t)nt * 16 * D_ + k2, D_, lane);
#pragma unroll
    for (int nt = 0; nt < 4; ++nt)
      acc[nt] = WMMA_F16(a1, b1[nt], acc[nt]);
  }

  // tail: k-steps 30 (in buf0) and 31
  a1 = load_a_frag(xrow + 31 * 32, D_, lane);
#pragma unroll
  for (int nt = 0; nt < 4; ++nt)
    b1[nt] = load_b_frag(wrow + (size_t)nt * 16 * D_ + 31 * 32, D_, lane);
#pragma unroll
  for (int nt = 0; nt < 4; ++nt)
    acc[nt] = WMMA_F16(a0, b0[nt], acc[nt]);
#pragma unroll
  for (int nt = 0; nt < 4; ++nt)
    acc[nt] = WMMA_F16(a1, b1[nt], acc[nt]);

  // C-frag layout: VGPR r -> M = r + (lane<16 ? 0 : 8); N = lane & 15.
  const int col  = lane & 15;
  const int roff = (lane < 16) ? 0 : 8;
#pragma unroll
  for (int nt = 0; nt < 4; ++nt) {
    const int c  = ctile + nt * 16 + col;
    const float bv = bias[c];
    const int h = c >> 6;      // head
    const int d = c & (DK_ - 1);
    if (!is_v) {
      // row-major [bh][t][dk]
#pragma unroll
      for (int r = 0; r < 8; ++r) {
        int n  = m0 + roff + r;
        int bb = n >> 11;            // n / T_
        int t  = n & (T_ - 1);
        outRM[((size_t)(bb * H_ + h) * T_ + t) * DK_ + d] = (_Float16)(acc[nt][r] + bv);
      }
    } else {
      // transposed [bh][dk][t]: rows r are contiguous along t -> one b128 store
      int n0 = m0 + roff;
      int bb = n0 >> 11;
      int t0 = n0 & (T_ - 1);
      v8h hv;
#pragma unroll
      for (int r = 0; r < 8; ++r) hv[r] = (_Float16)(acc[nt][r] + bv);
      *(v8h*)(outTR + ((size_t)(bb * H_ + h) * DK_ + d) * T_ + t0) = hv;
    }
  }
}

// ---------------------------------------------------------------------------
// Kernel 2: flash attention. Grid (32 q-tiles, 32 bh), block 128 (4 waves),
// each wave owns 16 q-rows. Online softmax, f32 accumulators, WMMA f16 for
// both S = Q*K^T and O += P*V. Reproduces the reference's mask-batch bug:
// mask batch index = bh % B; and the attn * qm multiply (zeroes masked q rows).
// V-fragment loads are hoisted above the softmax VALU section so their memory
// latency overlaps with exp/reduction math.
// ---------------------------------------------------------------------------
__global__ __launch_bounds__(128)
void mha_flash_attn(const _Float16* __restrict__ Qh, const _Float16* __restrict__ Kh,
                    const _Float16* __restrict__ Vt, const int* __restrict__ mask,
                    float* __restrict__ out) {
  __shared__ _Float16 lds_p[4][16][64];   // per-wave P tile, 8 KB total

  const int lane = threadIdx.x & 31;
  const int wave = threadIdx.x >> 5;
  const int bh   = blockIdx.y;
  const int q0   = blockIdx.x * 64 + wave * 16;
  const int b    = bh >> 4;          // bh / H_
  const int h    = bh & (H_ - 1);
  const int mb   = bh & (B_ - 1);    // faithful reference bug: mask batch = bh % B
  const int col  = lane & 15;
  const int roff = (lane < 16) ? 0 : 8;
  const float scale = 0.125f;        // 1/sqrt(64)

  const _Float16* qbase = Qh + ((size_t)bh * T_ + q0) * DK_;
  const _Float16* kbase = Kh + (size_t)bh * T_ * DK_;
  const _Float16* vbase = Vt + (size_t)bh * DK_ * T_;
  const int* mrow = mask + mb * T_;

  v16h aq0 = load_a_frag(qbase, DK_, lane);       // d = 0..31
  v16h aq1 = load_a_frag(qbase + 32, DK_, lane);  // d = 32..63

  v8f zero = {0.f, 0.f, 0.f, 0.f, 0.f, 0.f, 0.f, 0.f};
  v8f accO[4] = {zero, zero, zero, zero};
  float mrun[8], lrun[8];
#pragma unroll
  for (int r = 0; r < 8; ++r) { mrun[r] = -1e30f; lrun[r] = 0.f; }

  for (int kt = 0; kt < T_ / 64; ++kt) {
    const int k0 = kt * 64;

    if (kt + 1 < T_ / 64) {  // hint next K/V tiles into cache (global_prefetch_b8)
      __builtin_prefetch(kbase + (size_t)(k0 + 64 + lane * 2) * DK_, 0, 1);
      __builtin_prefetch(vbase + (size_t)(lane * 2) * T_ + k0 + 64, 0, 1);
    }

    int km[4];
#pragma unroll
    for (int nt = 0; nt < 4; ++nt) km[nt] = mrow[k0 + nt * 16 + col];

    // ---- S = Q * K^T : group all 8 K-fragment loads, then 8 WMMAs ----
    v16h bk0[4], bk1[4];
#pragma unroll
    for (int nt = 0; nt < 4; ++nt) {
      const _Float16* kp = kbase + (size_t)(k0 + nt * 16) * DK_;
      bk0[nt] = load_b_frag(kp, DK_, lane);
      bk1[nt] = load_b_frag(kp + 32, DK_, lane);
    }
    v8f s[4];
#pragma unroll
    for (int nt = 0; nt < 4; ++nt) {
      v8f t = WMMA_F16(aq0, bk0[nt], zero);
      s[nt] = WMMA_F16(aq1, bk1[nt], t);
    }

    // ---- issue V-fragment loads now; they complete during softmax math ----
    v16h bv0[4], bv1[4];
#pragma unroll
    for (int nt = 0; nt < 4; ++nt) {
      const _Float16* vp = vbase + (size_t)(nt * 16) * T_ + k0;
      bv0[nt] = load_b_frag(vp, T_, lane);        // kk 0..31
      bv1[nt] = load_b_frag(vp + 32, T_, lane);   // kk 32..63
    }

    // ---- scale + key mask, per-row max over 64 columns ----
    float pm[8];
#pragma unroll
    for (int r = 0; r < 8; ++r) pm[r] = -1e30f;
#pragma unroll
    for (int nt = 0; nt < 4; ++nt) {
#pragma unroll
      for (int r = 0; r < 8; ++r) {
        float v  = (km[nt] == 0) ? -1e9f : s[nt][r] * scale;
        s[nt][r] = v;
        pm[r]    = fmaxf(pm[r], v);
      }
    }
#pragma unroll
    for (int r = 0; r < 8; ++r) {   // rows live across 16-lane half groups
      float v = pm[r];
      v = fmaxf(v, __shfl_xor(v, 1));
      v = fmaxf(v, __shfl_xor(v, 2));
      v = fmaxf(v, __shfl_xor(v, 4));
      v = fmaxf(v, __shfl_xor(v, 8));
      pm[r] = v;
    }

    float alpha[8];
#pragma unroll
    for (int r = 0; r < 8; ++r) {
      float mnew = fmaxf(mrun[r], pm[r]);
      alpha[r]   = __expf(mrun[r] - mnew);
      mrun[r]    = mnew;
    }

    // ---- P = exp(S - m), spill to LDS as f16 (C-layout -> row-major) ----
    float ps[8];
#pragma unroll
    for (int r = 0; r < 8; ++r) ps[r] = 0.f;
#pragma unroll
    for (int nt = 0; nt < 4; ++nt) {
#pragma unroll
      for (int r = 0; r < 8; ++r) {
        float p = __expf(s[nt][r] - mrun[r]);
        ps[r] += p;
        lds_p[wave][roff + r][nt * 16 + col] = (_Float16)p;
      }
    }
#pragma unroll
    for (int r = 0; r < 8; ++r) {
      float v = ps[r];
      v += __shfl_xor(v, 1);
      v += __shfl_xor(v, 2);
      v += __shfl_xor(v, 4);
      v += __shfl_xor(v, 8);
      lrun[r] = lrun[r] * alpha[r] + v;
    }
#pragma unroll
    for (int nt = 0; nt < 4; ++nt)
#pragma unroll
      for (int r = 0; r < 8; ++r) accO[nt][r] *= alpha[r];

    // wave-private LDS store->load: DS ops are in-order per wave; keep the
    // compiler from reordering and make the dependency explicit.
    asm volatile("s_wait_dscnt 0" ::: "memory");

    v16h ap0 = load_a_frag(&lds_p[wave][0][0], 64, lane);   // kk = 0..31
    v16h ap1 = load_a_frag(&lds_p[wave][0][32], 64, lane);  // kk = 32..63

    // ---- O += P * V  (V stored transposed [d][t] => direct B-fragments) ----
#pragma unroll
    for (int nt = 0; nt < 4; ++nt) {
      accO[nt] = WMMA_F16(ap0, bv0[nt], accO[nt]);
      accO[nt] = WMMA_F16(ap1, bv1[nt], accO[nt]);
    }
  }

  // ---- epilogue: 1/l, q-row mask (reference's attn*qm), store f32 ----
  float inv[8];
#pragma unroll
  for (int r = 0; r < 8; ++r) {
    float qm = (float)mrow[q0 + roff + r];
    inv[r] = qm / lrun[r];
  }
#pragma unroll
  for (int nt = 0; nt < 4; ++nt) {
    const int d = h * DK_ + nt * 16 + col;
#pragma unroll
    for (int r = 0; r < 8; ++r) {
      const int t = q0 + roff + r;
      out[((size_t)b * T_ + t) * D_ + d] = accO[nt][r] * inv[r];
    }
  }
}

// ---------------------------------------------------------------------------
extern "C" void kernel_launch(void* const* d_in, const int* in_sizes, int n_in,
                              void* d_out, int out_size, void* d_ws, size_t ws_size,
                              hipStream_t stream) {
  (void)in_sizes; (void)n_in; (void)out_size; (void)ws_size;

  const float* queries = (const float*)d_in[0];
  const float* keys    = (const float*)d_in[1];
  const int*   keymask = (const int*)d_in[2];
  const float* Wq = (const float*)d_in[3];
  const float* bq = (const float*)d_in[4];
  const float* Wk = (const float*)d_in[5];
  const float* bk = (const float*)d_in[6];
  const float* Wv = (const float*)d_in[7];
  const float* bv = (const float*)d_in[8];
  float* out = (float*)d_out;

  const size_t NTD = (size_t)B_ * T_ * D_;   // 4,194,304
  const size_t DD  = (size_t)D_ * D_;        // 1,048,576

  _Float16* ws = (_Float16*)d_ws;
  _Float16* Qh = ws;
  _Float16* Kh = Qh + NTD;
  _Float16* Vt = Kh + NTD;
  _Float16* Xq = Vt + NTD;
  _Float16* Xk = Xq + NTD;
  _Float16* Wh = Xk + NTD;   // 3 * DD halves

  auto cvt = [stream](const float* s, _Float16* d, size_t n) {
    int n4 = (int)(n >> 2);
    mha_cvt_f32f16<<<(n4 + 255) / 256, 256, 0, stream>>>(s, d, n4);
  };
  cvt(queries, Xq, NTD);
  cvt(keys,    Xk, NTD);
  cvt(Wq, Wh,          DD);
  cvt(Wk, Wh + DD,     DD);
  cvt(Wv, Wh + 2 * DD, DD);

  dim3 pgrid(64, 16);   // 4096/64 M-tiles x 1024/64 N-tiles
  mha_qkv_proj<<<pgrid, 128, 0, stream>>>(Xq, Wh,          bq, Qh, nullptr, 0);
  mha_qkv_proj<<<pgrid, 128, 0, stream>>>(Xk, Wh + DD,     bk, Kh, nullptr, 0);
  mha_qkv_proj<<<pgrid, 128, 0, stream>>>(Xk, Wh + 2 * DD, bv, nullptr, Vt, 1);

  mha_flash_attn<<<dim3(32, 32), 128, 0, stream>>>(Qh, Kh, Vt, keymask, out);
}
